// critic_14156212208143
// MI455X (gfx1250) — matrix operand order repaired and checked by reading, not compiled, not run on hardware
//
#include <hip/hip_runtime.h>

typedef unsigned short u16;
typedef unsigned int   u32;
typedef __attribute__((ext_vector_type(16))) __bf16 v16bf;
typedef __attribute__((ext_vector_type(8)))  float  v8f;

struct Frag { uint4 lo, hi; };
static_assert(sizeof(Frag) == 32, "frag size");
static_assert(sizeof(v16bf) == 32, "v16bf size");

#define SEQ_  128
#define B_    256
#define OBS_  64
#define H_    1024
#define SB_   (SEQ_ * B_)

__device__ __forceinline__ u16 f2bf(float f) {
  u32 u = __builtin_bit_cast(u32, f);
  u32 r = (u + 0x7FFFu + ((u >> 16) & 1u)) >> 16;
  return (u16)r;
}
__device__ __forceinline__ float bf2f(u16 h) {
  u32 u = ((u32)h) << 16;
  return __builtin_bit_cast(float, u);
}
__device__ __forceinline__ float sigmoidf_(float x) {
  return 1.0f / (1.0f + __expf(-x));
}

// A-fragment: 16x32 bf16, lane holds row (lane&15), K runs [kb,kb+8) and [kb+16,kb+24)
__device__ __forceinline__ v16bf load_afrag(const u16* A, int lda, int row, int k) {
  const uint4* p = reinterpret_cast<const uint4*>(A + (size_t)row * lda + k);
  Frag f; f.lo = p[0]; f.hi = p[2];
  return __builtin_bit_cast(v16bf, f);
}
// masked A-fragment: zero whole fragment when this row's episode was reset
__device__ __forceinline__ v16bf load_afrag_mask(const u16* A, int lda, int row, int k, int drow) {
  const uint4* p = reinterpret_cast<const uint4*>(A + (size_t)row * lda + k);
  Frag f; f.lo = p[0]; f.hi = p[2];
  if (drow) { f.lo = make_uint4(0u, 0u, 0u, 0u); f.hi = make_uint4(0u, 0u, 0u, 0u); }
  return __builtin_bit_cast(v16bf, f);
}
// B-fragment: 32x16 bf16, lane holds column (lane&15), K runs [kb,kb+8) and [kb+16,kb+24)
__device__ __forceinline__ v16bf load_bfrag(const u16* W, int ldw, int row, int k) {
  const uint4* p = reinterpret_cast<const uint4*>(W + (size_t)row * ldw + k);
  Frag f; f.lo = p[0]; f.hi = p[2];
  return __builtin_bit_cast(v16bf, f);
}

// ---------------- device-wide sense-reversing barrier (persistent kernel) ---------------
__device__ __forceinline__ void grid_barrier(u32* cnt, u32* gen, int nblocks) {
  __syncthreads();
  if (threadIdx.x == 0) {
    u32 g0 = __hip_atomic_load(gen, __ATOMIC_RELAXED, __HIP_MEMORY_SCOPE_AGENT);
    u32 old = __hip_atomic_fetch_add(cnt, 1u, __ATOMIC_ACQ_REL, __HIP_MEMORY_SCOPE_AGENT);
    if (old == (u32)(nblocks - 1)) {
      __hip_atomic_store(cnt, 0u, __ATOMIC_RELAXED, __HIP_MEMORY_SCOPE_AGENT);
      __hip_atomic_store(gen, g0 + 1u, __ATOMIC_RELEASE, __HIP_MEMORY_SCOPE_AGENT);
    } else {
      while (__hip_atomic_load(gen, __ATOMIC_ACQUIRE, __HIP_MEMORY_SCOPE_AGENT) == g0) {
        __builtin_amdgcn_s_sleep(1);
      }
    }
  }
  __syncthreads();
}

// ---------------- one LSTM layer, one wave tile: 2 M-tiles x 16 cols x 4 gates ----------
// Every B-fragment is reused by two WMMAs (two M-tiles) -> halves L2 weight traffic and
// doubles WMMA issue density per wave.
__device__ __forceinline__ void lstm_layer_tile(
    const u16* __restrict__ X, int Kx,
    const u16* __restrict__ Hprev,
    const u16* __restrict__ Wih,
    const u16* __restrict__ Whh,
    const float bs[4],
    const int* __restrict__ done_t,
    float* __restrict__ cstate,
    u16* __restrict__ hout_b,
    float* __restrict__ hout_f, int write_f,
    int arow0, int akb, int bo, int bkb, int col, int rbase0) {
  v8f acc[2][4] = {};
  const int arow1 = arow0 + 16;

  // input contribution (no mask)
  for (int kc = 0; kc < Kx; kc += 32) {
    v16bf a0 = load_afrag(X, Kx, arow0, kc + akb);
    v16bf a1 = load_afrag(X, Kx, arow1, kc + akb);
#pragma unroll
    for (int g = 0; g < 4; ++g) {
      v16bf b = load_bfrag(Wih, Kx, g * H_ + bo, kc + bkb);
      acc[0][g] = __builtin_amdgcn_wmma_f32_16x16x32_bf16(
          false, a0, false, b, (short)0, acc[0][g], false, false);
      acc[1][g] = __builtin_amdgcn_wmma_f32_16x16x32_bf16(
          false, a1, false, b, (short)0, acc[1][g], false, false);
    }
  }

  // recurrent contribution with reset mask (done==1 -> zero state)
  const int drow0 = done_t[arow0];
  const int drow1 = done_t[arow1];
  for (int kc = 0; kc < H_; kc += 32) {
    v16bf a0 = load_afrag_mask(Hprev, H_, arow0, kc + akb, drow0);
    v16bf a1 = load_afrag_mask(Hprev, H_, arow1, kc + akb, drow1);
#pragma unroll
    for (int g = 0; g < 4; ++g) {
      v16bf b = load_bfrag(Whh, H_, g * H_ + bo, kc + bkb);
      acc[0][g] = __builtin_amdgcn_wmma_f32_16x16x32_bf16(
          false, a0, false, b, (short)0, acc[0][g], false, false);
      acc[1][g] = __builtin_amdgcn_wmma_f32_16x16x32_bf16(
          false, a1, false, b, (short)0, acc[1][g], false, false);
    }
  }

  // epilogue: PyTorch gate order i, f, g, o
#pragma unroll
  for (int tile = 0; tile < 2; ++tile) {
    const int rbase = rbase0 + tile * 16;
#pragma unroll
    for (int i = 0; i < 8; ++i) {
      const int r = rbase + i;
      const float gi = acc[tile][0][i] + bs[0];
      const float gf = acc[tile][1][i] + bs[1];
      const float gg = acc[tile][2][i] + bs[2];
      const float go = acc[tile][3][i] + bs[3];
      const float m  = done_t[r] ? 0.0f : 1.0f;
      const size_t idx = (size_t)r * H_ + col;
      const float cprev = cstate[idx] * m;
      const float c2 = sigmoidf_(gf) * cprev + sigmoidf_(gi) * tanhf(gg);
      const float h2 = sigmoidf_(go) * tanhf(c2);
      cstate[idx] = c2;
      hout_b[idx] = f2bf(h2);
      if (write_f) hout_f[idx] = h2;
    }
  }
}

// ---------------- persistent kernel: all 128 steps x 2 layers, grid-synced --------------
// Block = 256 threads = 8 waves (4M x 2N); wave owns 2 M-tiles (32 rows) x 16 cols x 4 gates.
// Block covers M=128 x N=32; grid = (2, 32) = 64 blocks (co-resident; barrier is safe).
__global__ __launch_bounds__(256)
void lstm_persistent_kernel(const u16* __restrict__ x_b,
                            const int* __restrict__ done,
                            u16* pp0, u16* pp1,
                            const u16* __restrict__ h1init,
                            u16* __restrict__ h1_all,
                            const u16* __restrict__ wih0,
                            const u16* __restrict__ whh0,
                            const u16* __restrict__ wih1,
                            const u16* __restrict__ whh1,
                            const float* __restrict__ bih0,
                            const float* __restrict__ bhh0,
                            const float* __restrict__ bih1,
                            const float* __restrict__ bhh1,
                            float* __restrict__ cs0,
                            float* __restrict__ cs1,
                            float* __restrict__ h0f,
                            float* __restrict__ h1f,
                            u32* bar) {
  const int lane   = threadIdx.x & 31;
  const int wv     = threadIdx.x >> 5;
  const int mw0    = blockIdx.x * 128 + (wv & 3) * 32;   // wave's first row (2 tiles)
  const int n0     = blockIdx.y * 32 + (wv >> 2) * 16;
  const int arow0  = mw0 + (lane & 15);
  const int akb    = (lane >> 4) * 8;
  const int bo     = n0 + (lane & 15);
  const int bkb    = (lane >> 4) * 8;
  const int col    = n0 + (lane & 15);
  const int rbase0 = mw0 + (lane >> 4) * 8;
  const int nblocks = (int)(gridDim.x * gridDim.y);
  u32* bar_cnt = bar;
  u32* bar_gen = bar + 64;  // separate cache lines

  float bsA[4], bsB[4];
#pragma unroll
  for (int g = 0; g < 4; ++g) {
    bsA[g] = bih0[g * H_ + col] + bhh0[g * H_ + col];
    bsB[g] = bih1[g * H_ + col] + bhh1[g * H_ + col];
  }

  // plain pointer swap (keeps loads in the global address space; no FLAT)
  u16* h_cur = pp0;
  u16* h_nxt = pp1;
  for (int t = 0; t < SEQ_; ++t) {
    const int* done_t = done + t * B_;
    const int wf = (t == SEQ_ - 1) ? 1 : 0;

    // layer 0: X = x_t (K=64), Hprev = h_cur -> h_nxt
    lstm_layer_tile(x_b + (size_t)t * B_ * OBS_, OBS_, h_cur, wih0, whh0,
                    bsA, done_t, cs0, h_nxt, h0f, wf,
                    arow0, akb, bo, bkb, col, rbase0);
    grid_barrier(bar_cnt, bar_gen, nblocks);

    // layer 1: X = h0_new (K=1024), Hprev = h1_all[t-1] (or h1init) -> h1_all[t]
    const u16* hp1 = t ? (h1_all + (size_t)(t - 1) * B_ * H_) : h1init;
    lstm_layer_tile(h_nxt, H_, hp1, wih1, whh1,
                    bsB, done_t, cs1, h1_all + (size_t)t * B_ * H_, h1f, wf,
                    arow0, akb, bo, bkb, col, rbase0);
    grid_barrier(bar_cnt, bar_gen, nblocks);

    u16* tmp = h_cur; h_cur = h_nxt; h_nxt = tmp;
  }
}

// ---------------- GEMM + bias + ELU (MLP layers), bf16 in/out, fp32 accumulate ----------
__global__ __launch_bounds__(256)
void gemm_elu_kernel(const u16* __restrict__ A, int K,
                     const u16* __restrict__ W,
                     const float* __restrict__ bias,
                     u16* __restrict__ out, int N) {
  const int lane = threadIdx.x & 31;
  const int wv   = threadIdx.x >> 5;
  const int m0   = blockIdx.x * 64 + (wv & 3) * 16;
  const int n0   = blockIdx.y * 32 + (wv >> 2) * 16;
  const int arow = m0 + (lane & 15);
  const int akb  = (lane >> 4) * 8;
  const int bo   = n0 + (lane & 15);
  const int bkb  = (lane >> 4) * 8;

  v8f acc = {};
  for (int kc = 0; kc < K; kc += 32) {
    v16bf a = load_afrag(A, K, arow, kc + akb);
    v16bf b = load_bfrag(W, K, bo, kc + bkb);
    acc = __builtin_amdgcn_wmma_f32_16x16x32_bf16(
        false, a, false, b, (short)0, acc, false, false);
  }
  const int col = n0 + (lane & 15);
  const float bc = bias[col];
  const int rbase = m0 + (lane >> 4) * 8;
#pragma unroll
  for (int i = 0; i < 8; ++i) {
    float v = acc[i] + bc;
    v = (v > 0.0f) ? v : (__expf(v) - 1.0f);
    out[(size_t)(rbase + i) * N + col] = f2bf(v);
  }
}

// ---------------- LayerNorm in place over bf16 hidden (one wave32 per row) -------------
__global__ __launch_bounds__(256)
void ln_kernel(u16* __restrict__ h, const float* __restrict__ g,
               const float* __restrict__ b) {
  const int lane = threadIdx.x & 31;
  const int row  = blockIdx.x * 8 + (threadIdx.x >> 5);
  u16* p = h + (size_t)row * H_;
  float s = 0.0f, ss = 0.0f;
  for (int k = lane; k < H_; k += 32) {
    const float v = bf2f(p[k]);
    s += v; ss += v * v;
  }
#pragma unroll
  for (int off = 16; off > 0; off >>= 1) {
    s  += __shfl_xor(s, off, 32);
    ss += __shfl_xor(ss, off, 32);
  }
  const float mean = s * (1.0f / H_);
  const float var  = ss * (1.0f / H_) - mean * mean;
  const float rstd = rsqrtf(var + 1e-5f);
  for (int k = lane; k < H_; k += 32) {
    const float v = bf2f(p[k]);
    p[k] = f2bf((v - mean) * rstd * g[k] + b[k]);
  }
}

// ---------------- final value head: per-row 64-dot --------------------------------------
__global__ __launch_bounds__(256)
void value_kernel(const u16* __restrict__ a3, const u16* __restrict__ wv,
                  const float* __restrict__ bv, float* __restrict__ out) {
  const int r = blockIdx.x * blockDim.x + threadIdx.x;
  float s = bv[0];
  const u16* row = a3 + (size_t)r * 64;
#pragma unroll
  for (int k = 0; k < 64; ++k) s += bf2f(row[k]) * bf2f(wv[k]);
  out[r] = s;
}

// ---------------- fp32 -> bf16 convert ---------------------------------------------------
__global__ __launch_bounds__(256)
void cvt_bf16_kernel(const float* __restrict__ in, u16* __restrict__ out, int n) {
  const int i = blockIdx.x * blockDim.x + threadIdx.x;
  if (i < n) out[i] = f2bf(in[i]);
}

// ---------------- barrier state init (ws is poisoned; must re-init every call) ----------
__global__ void bar_init_kernel(u32* bar) {
  bar[0]  = 0u;   // cnt
  bar[64] = 0u;   // gen
}

extern "C" void kernel_launch(void* const* d_in, const int* in_sizes, int n_in,
                              void* d_out, int out_size, void* d_ws, size_t ws_size,
                              hipStream_t stream) {
  const float* x    = (const float*)d_in[0];
  const int*   done = (const int*)  d_in[1];
  const float* h0   = (const float*)d_in[2];
  const float* c0   = (const float*)d_in[3];
  const float* Wih0 = (const float*)d_in[4];
  const float* Whh0 = (const float*)d_in[5];
  const float* bih0 = (const float*)d_in[6];
  const float* bhh0 = (const float*)d_in[7];
  const float* Wih1 = (const float*)d_in[8];
  const float* Whh1 = (const float*)d_in[9];
  const float* bih1 = (const float*)d_in[10];
  const float* bhh1 = (const float*)d_in[11];
  const float* lng  = (const float*)d_in[12];
  const float* lnb  = (const float*)d_in[13];
  const float* W1   = (const float*)d_in[14];
  const float* b1   = (const float*)d_in[15];
  const float* W2   = (const float*)d_in[16];
  const float* b2   = (const float*)d_in[17];
  const float* W3   = (const float*)d_in[18];
  const float* b3   = (const float*)d_in[19];
  const float* Wv   = (const float*)d_in[20];
  const float* bv   = (const float*)d_in[21];
  float* out = (float*)d_out;

  char* ws = (char*)d_ws;
  auto alloc = [&](size_t bytes) -> char* {
    char* p = ws;
    ws += (bytes + 255) & ~(size_t)255;
    return p;
  };
  u16* x_b    = (u16*)alloc((size_t)SB_ * OBS_ * 2);
  u16* wih0_b = (u16*)alloc((size_t)4 * H_ * OBS_ * 2);
  u16* whh0_b = (u16*)alloc((size_t)4 * H_ * H_ * 2);
  u16* wih1_b = (u16*)alloc((size_t)4 * H_ * H_ * 2);
  u16* whh1_b = (u16*)alloc((size_t)4 * H_ * H_ * 2);
  u16* w1_b   = (u16*)alloc((size_t)512 * H_ * 2);
  u16* w2_b   = (u16*)alloc((size_t)128 * 512 * 2);
  u16* w3_b   = (u16*)alloc((size_t)64 * 128 * 2);
  u16* wv_b   = (u16*)alloc((size_t)64 * 2);
  u16* pp0    = (u16*)alloc((size_t)B_ * H_ * 2);
  u16* pp1    = (u16*)alloc((size_t)B_ * H_ * 2);
  u16* h1init = (u16*)alloc((size_t)B_ * H_ * 2);
  u16* h1_all = (u16*)alloc((size_t)SB_ * H_ * 2);
  float* cs0  = (float*)alloc((size_t)B_ * H_ * 4);
  float* cs1  = (float*)alloc((size_t)B_ * H_ * 4);
  float* h0f  = (float*)alloc((size_t)B_ * H_ * 4);
  float* h1f  = (float*)alloc((size_t)B_ * H_ * 4);
  u16* a1     = (u16*)alloc((size_t)SB_ * 512 * 2);
  u16* a2     = (u16*)alloc((size_t)SB_ * 128 * 2);
  u16* a3     = (u16*)alloc((size_t)SB_ * 64 * 2);
  u32* bar    = (u32*)alloc(512);

  auto cvt = [&](const float* src, u16* dst, int n) {
    cvt_bf16_kernel<<<(n + 255) / 256, 256, 0, stream>>>(src, dst, n);
  };
  cvt(x,    x_b,    SB_ * OBS_);
  cvt(Wih0, wih0_b, 4 * H_ * OBS_);
  cvt(Whh0, whh0_b, 4 * H_ * H_);
  cvt(Wih1, wih1_b, 4 * H_ * H_);
  cvt(Whh1, whh1_b, 4 * H_ * H_);
  cvt(W1,   w1_b,   512 * H_);
  cvt(W2,   w2_b,   128 * 512);
  cvt(W3,   w3_b,   64 * 128);
  cvt(Wv,   wv_b,   64);
  cvt(h0,            pp0,    B_ * H_);  // layer-0 initial h
  cvt(h0 + B_ * H_,  h1init, B_ * H_);  // layer-1 initial h
  hipMemcpyAsync(cs0, c0,            (size_t)B_ * H_ * 4, hipMemcpyDeviceToDevice, stream);
  hipMemcpyAsync(cs1, c0 + B_ * H_,  (size_t)B_ * H_ * 4, hipMemcpyDeviceToDevice, stream);
  bar_init_kernel<<<1, 1, 0, stream>>>(bar);

  // all 128 steps x 2 layers in one persistent launch (64 blocks, grid-synced)
  lstm_persistent_kernel<<<dim3(2, 32), 256, 0, stream>>>(
      x_b, done, pp0, pp1, h1init, h1_all,
      wih0_b, whh0_b, wih1_b, whh1_b,
      bih0, bhh0, bih1, bhh1,
      cs0, cs1, h0f, h1f, bar);

  // outputs: [value (32768)] [hN (2*B*H)] [cN (2*B*H)]
  hipMemcpyAsync(out + SB_,               h0f, (size_t)B_ * H_ * 4, hipMemcpyDeviceToDevice, stream);
  hipMemcpyAsync(out + SB_ + 1 * B_ * H_, h1f, (size_t)B_ * H_ * 4, hipMemcpyDeviceToDevice, stream);
  hipMemcpyAsync(out + SB_ + 2 * B_ * H_, cs0, (size_t)B_ * H_ * 4, hipMemcpyDeviceToDevice, stream);
  hipMemcpyAsync(out + SB_ + 3 * B_ * H_, cs1, (size_t)B_ * H_ * 4, hipMemcpyDeviceToDevice, stream);

  ln_kernel<<<SB_ / 8, 256, 0, stream>>>(h1_all, lng, lnb);
  gemm_elu_kernel<<<dim3(SB_ / 64, 512 / 32), 256, 0, stream>>>(h1_all, H_,  w1_b, b1, a1, 512);
  gemm_elu_kernel<<<dim3(SB_ / 64, 128 / 32), 256, 0, stream>>>(a1,     512, w2_b, b2, a2, 128);
  gemm_elu_kernel<<<dim3(SB_ / 64, 64 / 32),  256, 0, stream>>>(a2,     128, w3_b, b3, a3, 64);
  value_kernel<<<SB_ / 256, 256, 0, stream>>>(a3, wv_b, bv, out);
}
